// BinaryLinear_22471268892739
// MI455X (gfx1250) — matrix operand order repaired and checked by reading, not compiled, not run on hardware
//
#include <hip/hip_runtime.h>
#include <hip/hip_bf16.h>

// ---------------------------------------------------------------------------
// BinaryLinear: y[32768,1024] = x[32768,1024] @ sign(W[1024,1024])^T + b[1024]
// bf16 WMMA with truncation-based hi/lo split of x: products with sign(W) in
// {-1,0,+1} are exact, hi+lo carries ~16 mantissa bits -> ~f32 accuracy.
// Conversion uses v_perm_b32/v_and/v_sub only (no slow RNE emulation).
// ---------------------------------------------------------------------------

typedef __attribute__((ext_vector_type(16))) __bf16 v16bf;
typedef __attribute__((ext_vector_type(8)))  float  v8f;

struct B32reg { uint4 a, b; };   // 32 bytes == one 16-element bf16 fragment

#define N_TOKENS     32768
#define IN_FEAT      1024
#define OUT_FEAT     1024
#define KCHUNK       64          // K elements staged in LDS per iteration
#define NCHUNKS      (IN_FEAT / KCHUNK)       // 16
#define LDS_PITCH    (KCHUNK * 2 + 16)        // 144 B: bank-conflict-free rows
#define TILE_N       128
#define TILE_M       128

// --------------------------- kernel 1: sign pack ---------------------------
// Bt[n*1024 + k] = sign(W[n][k]) as bf16 bit pattern (same layout as W).
__global__ __launch_bounds__(256) void sign_pack_kernel(
        const float* __restrict__ W, unsigned short* __restrict__ Bt) {
    const int i = blockIdx.x * 256 + threadIdx.x;    // 262144 threads, 4 elems each
    const float4 w = ((const float4*)W)[i];
    unsigned s0 = (w.x > 0.f) ? 0x3F80u : ((w.x < 0.f) ? 0xBF80u : 0u);
    unsigned s1 = (w.y > 0.f) ? 0x3F80u : ((w.y < 0.f) ? 0xBF80u : 0u);
    unsigned s2 = (w.z > 0.f) ? 0x3F80u : ((w.z < 0.f) ? 0xBF80u : 0u);
    unsigned s3 = (w.w > 0.f) ? 0x3F80u : ((w.w < 0.f) ? 0xBF80u : 0u);
    ((uint2*)Bt)[i] = make_uint2(s0 | (s1 << 16), s2 | (s3 << 16));
}

// --------------------------- kernel 2: GEMM --------------------------------
__global__ __launch_bounds__(256) void binlin_gemm_kernel(
        const float* __restrict__ x,
        const unsigned short* __restrict__ Bt,
        const float* __restrict__ bias,
        float* __restrict__ out) {

    __shared__ __align__(16) unsigned char ldsB[2][TILE_N * LDS_PITCH]; // 2x18KB

    const int tid    = threadIdx.x;
    const int wave   = tid >> 5;
    const int lane   = tid & 31;
    const int lane16 = lane & 15;
    const int half   = lane >> 4;           // 0: lanes 0-15, 1: lanes 16-31

    const int nblk  = blockIdx.x & 7;       // 8 N-blocks of 128
    const int mblk  = blockIdx.x >> 3;      // 256 M-blocks of 128
    const int nbase = nblk * TILE_N;
    const int mbase = mblk * TILE_M;

    // ---- per-wave A addressing (16-bit A 16x32 VGPR layout) ----
    const int m_row = mbase + wave * 16 + lane16;
    const float* xrow = x + (size_t)m_row * IN_FEAT;
    const int kb = half * 16;               // K base within a 32-wide step

    // ---- accumulators: 8 N-tiles of 16x16, bias-initialized ----
    // C layout: lane holds column n = ntile*16 + lane16 in all 8 VGPRs.
    v8f acc[8];
#pragma unroll
    for (int nt = 0; nt < 8; ++nt) {
        float b = bias[nbase + nt * 16 + lane16];
        v8f v = { b, b, b, b, b, b, b, b };
        acc[nt] = v;
    }

    // ---- cooperative B staging: 128 rows(n) x 64 cols(k) bf16 per chunk ----
    const int srow = tid >> 1;              // 0..127
    const int sh64 = tid & 1;               // which 32-element half of the row
    const unsigned short* bt_row = Bt + (size_t)(nbase + srow) * IN_FEAT + sh64 * 32;
    uint4 breg0, breg1, breg2, breg3;

    {   // prime the pipeline: chunk 0 -> registers
        const uint4* g = (const uint4*)(bt_row);
        breg0 = g[0]; breg1 = g[1]; breg2 = g[2]; breg3 = g[3];
    }

    for (int kcb = 0; kcb < NCHUNKS; ++kcb) {
        const int buf = kcb & 1;
        const int k0  = kcb * KCHUNK;

        // registers -> LDS for current chunk
        {
            uint4* p = (uint4*)(&ldsB[buf][srow * LDS_PITCH + sh64 * 64]);
            p[0] = breg0; p[1] = breg1; p[2] = breg2; p[3] = breg3;
        }
        __syncthreads();

        // issue next chunk's global loads (overlap with compute below)
        if (kcb + 1 < NCHUNKS) {
            const uint4* g = (const uint4*)(bt_row + (kcb + 1) * KCHUNK);
            breg0 = g[0]; breg1 = g[1]; breg2 = g[2]; breg3 = g[3];
            // pull the next A chunk toward the WGP caches as well
            __builtin_prefetch(xrow + (kcb + 1) * KCHUNK, 0, 3);
        }

        const unsigned char* lb = ldsB[buf];

#pragma unroll
        for (int ks = 0; ks < KCHUNK; ks += 32) {
            // ---- load 16 f32 of x for this lane, per A 16x32 layout ----
            const float* ap = xrow + k0 + ks + kb;
            float4 f0 = *(const float4*)(ap + 0);
            float4 f1 = *(const float4*)(ap + 4);
            float4 f2 = *(const float4*)(ap + 8);
            float4 f3 = *(const float4*)(ap + 12);
            float fa[16] = { f0.x, f0.y, f0.z, f0.w, f1.x, f1.y, f1.z, f1.w,
                             f2.x, f2.y, f2.z, f2.w, f3.x, f3.y, f3.z, f3.w };

            // ---- truncation split: hi = top 16 bits, lo = bf16(e - hi) ----
            // pack pairs with v_perm_b32; residual is exact in f32.
            unsigned hw[8], lw[8];
#pragma unroll
            for (int j = 0; j < 8; ++j) {
                float e0 = fa[2 * j], e1 = fa[2 * j + 1];
                unsigned u0 = __float_as_uint(e0);
                unsigned u1 = __float_as_uint(e1);
                // dst = { hi16(e1), hi16(e0) } in one v_perm_b32
                hw[j] = __builtin_amdgcn_perm(u1, u0, 0x07060302u);
                float h0 = __uint_as_float(u0 & 0xffff0000u);
                float h1 = __uint_as_float(u1 & 0xffff0000u);
                float r0 = e0 - h0;          // exact
                float r1 = e1 - h1;          // exact
                lw[j] = __builtin_amdgcn_perm(__float_as_uint(r1),
                                              __float_as_uint(r0), 0x07060302u);
            }
            B32reg H = { make_uint4(hw[0], hw[1], hw[2], hw[3]),
                         make_uint4(hw[4], hw[5], hw[6], hw[7]) };
            B32reg L = { make_uint4(lw[0], lw[1], lw[2], lw[3]),
                         make_uint4(lw[4], lw[5], lw[6], lw[7]) };
            v16bf ah = __builtin_bit_cast(v16bf, H);
            v16bf al = __builtin_bit_cast(v16bf, L);

            // ---- 8 N-tiles: B fragment from LDS + 2 WMMAs (hi, lo) ----
#pragma unroll
            for (int nt = 0; nt < 8; ++nt) {
                // B 32x16 layout: lane -> n = lane16, k = half*16 + j (contig)
                const uint4* bp = (const uint4*)(lb
                        + (nt * 16 + lane16) * LDS_PITCH
                        + (ks + half * 16) * 2);
                B32reg BB = { bp[0], bp[1] };
                v16bf bv = __builtin_bit_cast(v16bf, BB);

                acc[nt] = __builtin_amdgcn_wmma_f32_16x16x32_bf16(
                        false, ah, false, bv, (short)0, acc[nt], false, false);
                acc[nt] = __builtin_amdgcn_wmma_f32_16x16x32_bf16(
                        false, al, false, bv, (short)0, acc[nt], false, false);
            }
        }
        __syncthreads();
    }

    // ---- store: lane's VGPR r holds (m = wave*16 + half*8 + r, n = nt*16+lane16)
    const int m0 = mbase + wave * 16 + half * 8;
#pragma unroll
    for (int nt = 0; nt < 8; ++nt) {
        float* op = out + (size_t)m0 * OUT_FEAT + nbase + nt * 16 + lane16;
#pragma unroll
        for (int r = 0; r < 8; ++r) {
            op[(size_t)r * OUT_FEAT] = acc[nt][r];
        }
    }
}

// --------------------------- launcher --------------------------------------
extern "C" void kernel_launch(void* const* d_in, const int* in_sizes, int n_in,
                              void* d_out, int out_size, void* d_ws, size_t ws_size,
                              hipStream_t stream) {
    const float* x    = (const float*)d_in[0];
    const float* W    = (const float*)d_in[1];
    const float* bias = (const float*)d_in[2];
    float* out = (float*)d_out;

    unsigned short* Bt = (unsigned short*)d_ws;   // 1024*1024 bf16 = 2 MB

    // Kernel 1: W -> bf16 sign matrix (4 elems/thread over 1M elems)
    sign_pack_kernel<<<(OUT_FEAT * IN_FEAT / 4 + 255) / 256, 256, 0, stream>>>(W, Bt);

    // Kernel 2: tiled WMMA GEMM, 128x128 tile per workgroup
    const int grid = (N_TOKENS / TILE_M) * (OUT_FEAT / TILE_N);   // 256*8 = 2048
    binlin_gemm_kernel<<<grid, 256, 0, stream>>>(x, Bt, bias, out);
}
